// Capsules_52862457479578
// MI455X (gfx1250) — matrix operand order repaired and checked by reading, not compiled, not run on hardware
//
#include <hip/hip_runtime.h>
#include <hip/hip_bf16.h>
#include <stdint.h>

// Problem constants (match reference)
static constexpr int BS = 32, CI = 1024, NI = 32, CO = 64, NO = 32;
static constexpr int M = CO * NO;          // 2048
static constexpr int ITERS = 3;

typedef __attribute__((ext_vector_type(16))) _Float16 v16h;
typedef __attribute__((ext_vector_type(8)))  float    v8f;

// quantize to 2^-bits grid, round-half-even (matches jnp.round)
__device__ __forceinline__ float qz(float x, float s, float inv_s) {
    return rintf(x * s) * inv_s;
}

__device__ __forceinline__ float wave_max32(float v) {
    #pragma unroll
    for (int off = 16; off > 0; off >>= 1) v = fmaxf(v, __shfl_xor(v, off, 32));
    return v;
}
__device__ __forceinline__ float wave_sum32(float v) {
    #pragma unroll
    for (int off = 16; off > 0; off >>= 1) v += __shfl_xor(v, off, 32);
    return v;
}

__device__ __forceinline__ void unpack2(unsigned u, float s, float& a, float& b) {
    a = (float)((short)(u & 0xffffu)) * s;
    b = (float)((short)(u >> 16)) * s;
}

// -------------------------------------------------------------------------
// Kernel 1: votes[b,i,m] = quantize(sum_n x[b,i,n]*w[i,n,m]) stored as i16
// (value * 2^qbits). One block per (i, 128-column slab); 8 waves, each wave
// one 16-col tile, two M-halves (b=0..15, 16..31) => 2 WMMAs/wave.
// -------------------------------------------------------------------------
__global__ __launch_bounds__(256) void votes_wmma_kernel(
    const float* __restrict__ x, const float* __restrict__ w,
    const int* __restrict__ qbits, short* __restrict__ votes)
{
    const int i    = blockIdx.y;
    const int mb   = blockIdx.x;              // 0..15
    const int tid  = threadIdx.x;
    const int lane = tid & 31, wv = tid >> 5;

    __shared__ float xs[BS][NI];              // x[:, i, :], 4 KB
    {
        const int b = tid >> 3, n4 = (tid & 7) * 4;
        const float4 v = *(const float4*)(x + ((size_t)b * CI + i) * NI + n4);
        xs[b][n4 + 0] = v.x; xs[b][n4 + 1] = v.y;
        xs[b][n4 + 2] = v.z; xs[b][n4 + 3] = v.w;
    }
    __syncthreads();

    const float scale = exp2f((float)qbits[0]);
    const int mcol  = mb * 128 + wv * 16 + (lane & 15);
    const int khalf = lane >> 4;              // 0: lanes 0-15, 1: lanes 16-31

    // B tile: 32x16 f16, K along VGPRs (2/half-pair), N = lane&15,
    // lanes 0-15 hold K=0..15, lanes 16-31 hold K=16..31.
    v16h bv;
    const float* wrow = w + (size_t)i * NI * M + mcol;
    #pragma unroll
    for (int j = 0; j < 8; ++j) {
        const int k0 = 2 * j + khalf * 16;
        bv[2 * j]     = (_Float16)wrow[(size_t)k0 * M];
        bv[2 * j + 1] = (_Float16)wrow[(size_t)(k0 + 1) * M];
    }

    #pragma unroll
    for (int h = 0; h < 2; ++h) {
        // A tile: 16x32 f16. lanes 0-15: K 0..15; lanes 16-31: K 16..31
        v16h av;
        const int brow = h * 16 + (lane & 15);
        const int kb   = khalf * 16;
        #pragma unroll
        for (int e = 0; e < 8; ++e) {
            av[e]     = (_Float16)xs[brow][kb + e];
            av[e + 8] = (_Float16)xs[brow][kb + 8 + e];
        }
        v8f c = {};
        c = __builtin_amdgcn_wmma_f32_16x16x32_f16(false, av, false, bv,
                                                   (short)0, c, false, false);
        // D: VGPR r -> row r (lanes 0-15) / r+8 (lanes 16-31), N = lane&15
        #pragma unroll
        for (int r = 0; r < 8; ++r) {
            const int bo = h * 16 + r + khalf * 8;
            float kq = rintf(c[r] * scale);
            kq = fminf(fmaxf(kq, -32768.f), 32767.f);
            votes[((size_t)bo * CI + i) * M + mcol] = (short)(int)kq;
        }
    }
}

// -------------------------------------------------------------------------
// Zero helper (graph-capture-safe scratch init)
// -------------------------------------------------------------------------
__global__ __launch_bounds__(256) void zero_kernel(float* __restrict__ p, int n) {
    int idx = blockIdx.x * 256 + threadIdx.x;
    if (idx < n) p[idx] = 0.f;
}

// -------------------------------------------------------------------------
// Kernel 2 (fused per iteration): for each (b,i) read votes row ONCE;
//   if use_act: dist[c] = <votes[b,i,c,:], act[b,c,:]>,
//               logits = qz(prev_logits + dist) (store)
//   route = qz(softmax(logits))   (stage 0: softmax of zeros -> uniform)
//   acc   += route[c] * votes[b,i,c,:]  -> block LDS partial -> global atomics
// Block: 256 thr (8 waves), covers b = blockIdx.y, i in [blockIdx.x*32, +32).
// Each wave handles 4 i's; each lane owns capsules c = 2*lane, 2*lane+1.
// -------------------------------------------------------------------------
__global__ __launch_bounds__(256) void route_fused_kernel(
    const short* __restrict__ votes, float* __restrict__ logits,
    const float* __restrict__ act_in, float* __restrict__ P,
    const int* __restrict__ qb, const int* __restrict__ qbr,
    int use_act, int use_prev)
{
    const int b = blockIdx.y;
    const int ibase = blockIdx.x * 32;
    const int tid = threadIdx.x, lane = tid & 31, wv = tid >> 5;

    __shared__ float act_s[CO * NO];   // 8 KB
    __shared__ float P_loc[CO * NO];   // 8 KB

    for (int e = tid; e < CO * NO; e += 256) P_loc[e] = 0.f;
    if (use_act)
        for (int e = tid; e < CO * NO; e += 256)
            act_s[e] = act_in[(size_t)b * (CO * NO) + e];
    __syncthreads();

    const float sq  = exp2f((float)qb[0]),  isq = 1.f / sq;
    const float sr  = exp2f((float)qbr[0]), isr = 1.f / sr;
    const float ivs = isq;                  // votes i16 -> float scale

    const int c0 = lane * 2;
    float acc[64];
    #pragma unroll
    for (int n = 0; n < 64; ++n) acc[n] = 0.f;

    for (int q = 0; q < 4; ++q) {
        const int i = ibase + wv * 4 + q;
        const uint4* row = (const uint4*)(votes + ((size_t)b * CI + i) * M);
        if (q < 3)  // prefetch next votes row -> global_prefetch_b8
            __builtin_prefetch((const void*)(votes + ((size_t)b * CI + i + 1) * M + lane * 64), 0, 1);

        uint4 vr[8];                        // lane's 64 i16: c0 rows then c0+1
        #pragma unroll
        for (int r = 0; r < 8; ++r) vr[r] = row[lane * 8 + r];

        float nl0 = 0.f, nl1 = 0.f;
        if (use_act) {
            float d0 = 0.f, d1 = 0.f;
            #pragma unroll
            for (int r = 0; r < 4; ++r) {
                const unsigned u0[4] = {vr[r].x, vr[r].y, vr[r].z, vr[r].w};
                const unsigned u1[4] = {vr[r + 4].x, vr[r + 4].y, vr[r + 4].z, vr[r + 4].w};
                #pragma unroll
                for (int cc = 0; cc < 4; ++cc) {
                    const int n = r * 8 + cc * 2;
                    float a, bb;
                    unpack2(u0[cc], ivs, a, bb);
                    d0 = fmaf(a,  act_s[c0 * 32 + n],     d0);
                    d0 = fmaf(bb, act_s[c0 * 32 + n + 1], d0);
                    unpack2(u1[cc], ivs, a, bb);
                    d1 = fmaf(a,  act_s[(c0 + 1) * 32 + n],     d1);
                    d1 = fmaf(bb, act_s[(c0 + 1) * 32 + n + 1], d1);
                }
            }
            const size_t lidx = ((size_t)b * CI + i) * CO + c0;
            float l0 = use_prev ? logits[lidx]     : 0.f;
            float l1 = use_prev ? logits[lidx + 1] : 0.f;
            nl0 = qz(l0 + d0, sr, isr);
            nl1 = qz(l1 + d1, sr, isr);
            logits[lidx]     = nl0;
            logits[lidx + 1] = nl1;
        }
        // softmax over 64 capsules (2 per lane) + quantize
        const float mx = wave_max32(fmaxf(nl0, nl1));
        const float e0 = expf(nl0 - mx), e1 = expf(nl1 - mx);
        const float sm = wave_sum32(e0 + e1);
        const float r0 = qz(e0 / sm, sq, isq);
        const float r1 = qz(e1 / sm, sq, isq);

        #pragma unroll
        for (int r = 0; r < 4; ++r) {
            const unsigned u0[4] = {vr[r].x, vr[r].y, vr[r].z, vr[r].w};
            const unsigned u1[4] = {vr[r + 4].x, vr[r + 4].y, vr[r + 4].z, vr[r + 4].w};
            #pragma unroll
            for (int cc = 0; cc < 4; ++cc) {
                const int n = r * 8 + cc * 2;
                float a, bb;
                unpack2(u0[cc], ivs, a, bb);
                acc[n]      = fmaf(r0, a,  acc[n]);
                acc[n + 1]  = fmaf(r0, bb, acc[n + 1]);
                unpack2(u1[cc], ivs, a, bb);
                acc[32 + n]     = fmaf(r1, a,  acc[32 + n]);
                acc[32 + n + 1] = fmaf(r1, bb, acc[32 + n + 1]);
            }
        }
    }

    #pragma unroll
    for (int n = 0; n < 32; ++n) {
        atomicAdd(&P_loc[c0 * 32 + n],       acc[n]);        // ds_add_f32
        atomicAdd(&P_loc[(c0 + 1) * 32 + n], acc[32 + n]);
    }
    __syncthreads();
    for (int e = tid; e < CO * NO; e += 256)
        atomicAdd(&P[(size_t)b * (CO * NO) + e], P_loc[e]);  // global_atomic_add_f32
}

// -------------------------------------------------------------------------
// Kernel 3: preactivate = qz(P + bias, r-bits); squash over n; act = qz(...)
// One wave per (b,c); lane = n.
// -------------------------------------------------------------------------
__global__ __launch_bounds__(256) void squash_kernel(
    const float* __restrict__ P, const float* __restrict__ bias,
    const int* __restrict__ qb, const int* __restrict__ qbr,
    float* __restrict__ act_out, float* __restrict__ out, int write_out)
{
    const int tid = threadIdx.x, lane = tid & 31, wv = tid >> 5;
    const int gcap = blockIdx.x * 8 + wv;       // b*64 + c
    const int c = gcap & 63;
    const float sq = exp2f((float)qb[0]),  isq = 1.f / sq;
    const float sr = exp2f((float)qbr[0]), isr = 1.f / sr;

    const size_t idx = (size_t)gcap * NO + lane;
    float p = P[idx] + bias[c * NO + lane];
    p = qz(p, sr, isr);
    const float n2  = wave_sum32(p * p);
    const float nrm = sqrtf(n2);
    float a = p * nrm / (1.f + n2);
    a = qz(a, sq, isq);
    act_out[idx] = a;
    if (write_out) out[idx] = a;
}

// -------------------------------------------------------------------------
extern "C" void kernel_launch(void* const* d_in, const int* in_sizes, int n_in,
                              void* d_out, int out_size, void* d_ws, size_t ws_size,
                              hipStream_t stream) {
    const float* x    = (const float*)d_in[0];   // [32,1024,32]
    const float* w    = (const float*)d_in[1];   // [1024,32,2048]
    const float* bias = (const float*)d_in[2];   // [64,32]
    const int*   qb   = (const int*)d_in[3];
    const int*   qbr  = (const int*)d_in[4];
    float* out = (float*)d_out;                  // [32,64,32]

    char* ws = (char*)d_ws;
    short* votes = (short*)ws;                                   // 128 MB i16
    size_t off = (size_t)BS * CI * M * sizeof(short);
    float* logits = (float*)(ws + off); off += (size_t)BS * CI * CO * sizeof(float); // 8 MB
    float* P      = (float*)(ws + off); off += (size_t)BS * CO * NO * sizeof(float); // 256 KB
    float* act    = (float*)(ws + off);                                              // 256 KB

    votes_wmma_kernel<<<dim3(16, CI), 256, 0, stream>>>(x, w, qb, votes);

    const int Pn = BS * CO * NO;
    for (int t = 0; t < ITERS; ++t) {
        zero_kernel<<<(Pn + 255) / 256, 256, 0, stream>>>(P, Pn);
        route_fused_kernel<<<dim3(CI / 32, BS), 256, 0, stream>>>(
            votes, logits, act, P, qb, qbr, /*use_act=*/t > 0, /*use_prev=*/t > 1);
        squash_kernel<<<(BS * CO) / 8, 256, 0, stream>>>(
            P, bias, qb, qbr, act, out, /*write_out=*/t == ITERS - 1);
    }
}
